// RNN_26242250178595
// MI455X (gfx1250) — compile-verified, matrix operand order checked by reading, not run on hardware
//
#include <hip/hip_runtime.h>
#include <hip/hip_bf16.h>

typedef __attribute__((ext_vector_type(16))) _Float16 v16h;
typedef __attribute__((ext_vector_type(8)))  _Float16 v8h;
typedef __attribute__((ext_vector_type(8)))  float    v8f;
typedef int v4i __attribute__((vector_size(16)));   // matches async-LDS builtin params

namespace {
constexpr int kB     = 16384;
constexpr int kT     = 10;
constexpr int kH     = 256;
constexpr int kG     = 4 * kH;   // 1024 gate columns
constexpr int kINr   = 39;       // real input feature count
constexpr int kIN0   = 64;       // padded to multiple of 32 for WMMA K
constexpr int kNOUT  = 500;
constexpr int kNOUTP = 512;
constexpr int kMT    = 32;       // batch rows per block (2 M-subtiles per wave)
}

#if __has_builtin(__builtin_amdgcn_global_load_async_to_lds_b128)
#define HAS_ASYNC_LDS 1
#else
#define HAS_ASYNC_LDS 0
#endif

__device__ __forceinline__ void async_wait_all() {
#if HAS_ASYNC_LDS
#if __has_builtin(__builtin_amdgcn_s_wait_asynccnt)
  __builtin_amdgcn_s_wait_asynccnt(0);
#else
  asm volatile("s_wait_asynccnt 0" ::: "memory");
#endif
#endif
}

__device__ __forceinline__ float sigmf(float x) { return 1.0f / (1.0f + __expf(-x)); }
__device__ __forceinline__ float tanhfast(float x) { return 2.0f / (1.0f + __expf(-2.0f * x)) - 1.0f; }
__device__ __forceinline__ v16h cat8(v8h lo, v8h hi) {
  return __builtin_shufflevector(lo, hi, 0,1,2,3,4,5,6,7,8,9,10,11,12,13,14,15);
}

// ---------------- Weight prep: Wcat[n][0:256)=Whh[n][:], Wcat[n][256:256+kin_pad)=Wih[n][:] (f16), bsum=bih+bhh
__global__ void prep_wcat_k(const float* __restrict__ Whh, const float* __restrict__ Wih,
                            int kin_real, int kin_pad,
                            const float* __restrict__ bi, const float* __restrict__ bh,
                            _Float16* __restrict__ Wcat, float* __restrict__ bsum) {
  const int kt = kH + kin_pad;
  const int idx = blockIdx.x * blockDim.x + threadIdx.x;
  const int total = kG * kt;
  if (idx < total) {
    const int n = idx / kt, k = idx % kt;
    float v;
    if (k < kH) {
      v = Whh[(size_t)n * kH + k];
    } else {
      const int kk = k - kH;
      v = (kk < kin_real) ? Wih[(size_t)n * kin_real + kk] : 0.0f;
    }
    Wcat[idx] = (_Float16)v;
  }
  if (idx < kG) bsum[idx] = bi[idx] + bh[idx];
}

__global__ void prep_wout_k(const float* __restrict__ Wout, const float* __restrict__ bout,
                            _Float16* __restrict__ WoutP, float* __restrict__ boutP) {
  const int idx = blockIdx.x * blockDim.x + threadIdx.x;
  if (idx < kNOUTP * kH) {
    const int n = idx / kH, k = idx % kH;
    WoutP[idx] = (_Float16)((n < kNOUT) ? Wout[(size_t)n * kH + k] : 0.0f);
  }
  if (idx < kNOUTP) boutP[idx] = (idx < kNOUT) ? bout[idx] : 0.0f;
}

// ---------------- Embedding gather: x0[bt][0:64) f16, zero-padded past 39
__global__ void gather_embed_k(const int* __restrict__ car, const int* __restrict__ reg,
                               const int* __restrict__ poi, const int* __restrict__ wk,
                               const int* __restrict__ tm,
                               const float* __restrict__ ce, const float* __restrict__ re,
                               const float* __restrict__ pe, const float* __restrict__ we,
                               const float* __restrict__ te,
                               _Float16* __restrict__ x0) {
  const int bt = blockIdx.x * blockDim.x + threadIdx.x;
  if (bt >= kB * kT) return;
  _Float16* dst = x0 + (size_t)bt * kIN0;
  const float* s0 = ce + (size_t)car[bt] * 16;
  const float* s1 = re + (size_t)reg[bt] * 8;
  const float* s2 = pe + (size_t)poi[bt] * 4;
  const float* s3 = we + (size_t)wk[bt]  * 3;
  const float* s4 = te + (size_t)tm[bt]  * 8;
  int k = 0;
  #pragma unroll
  for (int i = 0; i < 16; i++) dst[k++] = (_Float16)s0[i];
  #pragma unroll
  for (int i = 0; i < 8; i++)  dst[k++] = (_Float16)s1[i];
  #pragma unroll
  for (int i = 0; i < 4; i++)  dst[k++] = (_Float16)s2[i];
  #pragma unroll
  for (int i = 0; i < 3; i++)  dst[k++] = (_Float16)s3[i];
  #pragma unroll
  for (int i = 0; i < 8; i++)  dst[k++] = (_Float16)s4[i];
  #pragma unroll
  for (; k < kIN0; k++) dst[k] = (_Float16)0.0f;
}

// Stage x_t for 32 batch rows into an LDS slot with async-to-LDS (fallback: sync copy).
template<int KIN>
__device__ __forceinline__ void stage_x(const _Float16* __restrict__ xin,
                                        _Float16 (*Xslot)[KIN],
                                        int bRow, int t, int tid) {
  constexpr int LPR = KIN / 8;  // 16-byte loads per row
  for (int i = tid; i < kMT * LPR; i += 256) {
    const int m = i / LPR;
    const int c = (i % LPR) * 8;
    const _Float16* g = xin + ((size_t)(bRow + m) * kT + t) * KIN + c;
    _Float16* l = &Xslot[m][c];
#if HAS_ASYNC_LDS
    __builtin_amdgcn_global_load_async_to_lds_b128(
        (__attribute__((address_space(1))) v4i*)g,
        (__attribute__((address_space(3))) v4i*)l, 0, 0);
#else
    *(v8h*)l = *(const v8h*)g;
#endif
  }
}

// ---------------- Persistent LSTM layer: block owns 32 batch rows across all T steps.
// Wave w owns hidden cols j in [32w, 32w+32) and computes ALL four gates for them,
// so cell state c lives in that wave's registers in WMMA accumulator layout.
// Each B fragment (weights) is reused for two M-subtiles -> half the L2 weight traffic.
template<int KIN, bool WRITE_ALL>
__global__ __launch_bounds__(256) void lstm_layer_k(
    const _Float16* __restrict__ xin,   // [B*T, KIN] f16
    const _Float16* __restrict__ Wcat,  // [1024, kH+KIN] f16
    const float*    __restrict__ bsum,  // [1024]
    _Float16*       __restrict__ hout)  // WRITE_ALL ? [B*T,256] : [B,256] f16
{
  constexpr int KT = kH + KIN;
  __shared__ __align__(16) _Float16 Hbuf[kMT][kH];       // h state (f16)
  __shared__ __align__(16) _Float16 Xbuf[2][kMT][KIN];   // double-buffered x_t

  const int tid   = threadIdx.x;
  const int lane  = tid & 31;
  const int wave  = tid >> 5;
  const int bRow  = blockIdx.x * kMT;
  const int mlan  = lane & 15;     // row (A) / col (B) within a 16-tile
  const int khalf = lane >> 4;     // which 8-wide K chunk pair this lane holds
  const int j0    = wave * 32;     // this wave's hidden-column base

  // h0 = 0
  for (int i = tid; i < kMT * kH; i += 256)
    Hbuf[i / kH][i % kH] = (_Float16)0.0f;

  // kick off async staging of x_0 into slot 0
  stage_x<KIN>(xin, Xbuf[0], bRow, 0, tid);

  // biases for the 8 gate subtiles: s = gate*2 + half, n = gate*256 + j0 + 16*half
  float bias[8];
  #pragma unroll
  for (int s = 0; s < 8; s++) {
    const int n = (s >> 1) * kH + j0 + 16 * (s & 1);
    bias[s] = bsum[n + mlan];
  }

  v8f zero = {};
  v8f cacc[2][2];                  // cell state: [M-subtile][j-subtile], acc layout
  cacc[0][0] = zero; cacc[0][1] = zero; cacc[1][0] = zero; cacc[1][1] = zero;

  for (int t = 0; t < kT; t++) {
    async_wait_all();       // x_t staged
    __syncthreads();        // + h(t-1) visible to all waves

    // overlap: stage x_{t+1} into the other slot during this step's GEMM
    if (t + 1 < kT) stage_x<KIN>(xin, Xbuf[(t + 1) & 1], bRow, t + 1, tid);
    const _Float16 (*Xs)[KIN] = Xbuf[t & 1];

    v8f acc[2][8];
    #pragma unroll
    for (int s = 0; s < 8; s++) { acc[0][s] = zero; acc[1][s] = zero; }

    #pragma unroll
    for (int kb = 0; kb < KT / 32; kb++) {
      const int k0 = kb * 32 + khalf * 8;
      v16h a0, a1;
      if (kb * 32 < kH) {   // compile-time after unroll: h region
        a0 = cat8(*(const v8h*)&Hbuf[mlan][k0],      *(const v8h*)&Hbuf[mlan][k0 + 16]);
        a1 = cat8(*(const v8h*)&Hbuf[mlan + 16][k0], *(const v8h*)&Hbuf[mlan + 16][k0 + 16]);
      } else {              // x region (double-buffered slot)
        const int c0 = k0 - kH;
        a0 = cat8(*(const v8h*)&Xs[mlan][c0],        *(const v8h*)&Xs[mlan][c0 + 16]);
        a1 = cat8(*(const v8h*)&Xs[mlan + 16][c0],   *(const v8h*)&Xs[mlan + 16][c0 + 16]);
      }
      #pragma unroll
      for (int s = 0; s < 8; s++) {
        const int n = (s >> 1) * kH + j0 + 16 * (s & 1) + mlan;
        const _Float16* brow = Wcat + (size_t)n * KT;
        v16h bf = cat8(*(const v8h*)&brow[k0], *(const v8h*)&brow[k0 + 16]);
        acc[0][s] = __builtin_amdgcn_wmma_f32_16x16x32_f16(
            false, a0, false, bf, (short)0, acc[0][s], false, false);
        acc[1][s] = __builtin_amdgcn_wmma_f32_16x16x32_f16(
            false, a1, false, bf, (short)0, acc[1][s], false, false);
      }
    }
    __syncthreads();   // all GEMM reads of Hbuf done before h overwrite

    // LSTM cell on this wave's 32 hidden columns (i,f,g,o all local)
    #pragma unroll
    for (int mt = 0; mt < 2; mt++) {
      #pragma unroll
      for (int half = 0; half < 2; half++) {
        v8f gi = acc[mt][0 + half], gf = acc[mt][2 + half];
        v8f gg = acc[mt][4 + half], go = acc[mt][6 + half];
        v8f cc = cacc[mt][half];
        #pragma unroll
        for (int r = 0; r < 8; r++) {
          const float iv = sigmf(gi[r] + bias[0 + half]);
          const float fv = sigmf(gf[r] + bias[2 + half]);
          const float gv = tanhfast(gg[r] + bias[4 + half]);
          const float ov = sigmf(go[r] + bias[6 + half]);
          const float cn = fv * cc[r] + iv * gv;
          const float hv = ov * tanhfast(cn);
          cc[r] = cn;
          const int m = r + (khalf << 3) + mt * 16;  // accumulator row mapping
          const int j = j0 + 16 * half + mlan;       // accumulator col mapping
          Hbuf[m][j] = (_Float16)hv;                 // h for next step's A matrix
          if (WRITE_ALL) {
            hout[((size_t)(bRow + m) * kT + t) * kH + j] = (_Float16)hv;
          } else if (t == kT - 1) {
            hout[(size_t)(bRow + m) * kH + j] = (_Float16)hv;
          }
        }
        cacc[mt][half] = cc;
      }
    }
    // next iteration's top barrier covers Hbuf visibility
  }
}

// ---------------- Output projection: [B,256] f16 @ [512,256] f16 -> f32 [B,500]
__global__ __launch_bounds__(256) void out_gemm_k(
    const _Float16* __restrict__ h2,     // [B, 256]
    const _Float16* __restrict__ WoutP,  // [512, 256]
    const float*    __restrict__ boutP,  // [512]
    float*          __restrict__ out)    // [B, 500]
{
  const int tid   = threadIdx.x;
  const int lane  = tid & 31;
  const int wave  = tid >> 5;
  const int bRow  = blockIdx.x * 16;
  const int mlan  = lane & 15;
  const int khalf = lane >> 4;
  const int nBase = wave * 64;           // 8 waves x 64 cols = 512

  v8f zero = {};
  v8f acc[4];
  #pragma unroll
  for (int s = 0; s < 4; s++) acc[s] = zero;

  const _Float16* arow = h2 + (size_t)(bRow + mlan) * kH;
  #pragma unroll
  for (int kb = 0; kb < kH / 32; kb++) {
    const int k0 = kb * 32 + khalf * 8;
    v16h a = cat8(*(const v8h*)&arow[k0], *(const v8h*)&arow[k0 + 16]);
    #pragma unroll
    for (int s = 0; s < 4; s++) {
      const int n = nBase + s * 16 + mlan;
      const _Float16* brow = WoutP + (size_t)n * kH;
      v16h bf = cat8(*(const v8h*)&brow[k0], *(const v8h*)&brow[k0 + 16]);
      acc[s] = __builtin_amdgcn_wmma_f32_16x16x32_f16(
          false, a, false, bf, (short)0, acc[s], false, false);
    }
  }
  #pragma unroll
  for (int s = 0; s < 4; s++) {
    #pragma unroll
    for (int r = 0; r < 8; r++) {
      const int m   = r + (khalf << 3);
      const int col = nBase + s * 16 + mlan;
      if (col < kNOUT)
        out[(size_t)(bRow + m) * kNOUT + col] = acc[s][r] + boutP[col];
    }
  }
}

extern "C" void kernel_launch(void* const* d_in, const int* in_sizes, int n_in,
                              void* d_out, int out_size, void* d_ws, size_t ws_size,
                              hipStream_t stream) {
  (void)in_sizes; (void)n_in; (void)out_size; (void)ws_size;
  const int*   car  = (const int*)d_in[0];
  const int*   reg  = (const int*)d_in[1];
  const int*   poi  = (const int*)d_in[2];
  const int*   wk   = (const int*)d_in[3];
  const int*   tm   = (const int*)d_in[4];
  const float* ce   = (const float*)d_in[5];
  const float* re   = (const float*)d_in[6];
  const float* pe   = (const float*)d_in[7];
  const float* we   = (const float*)d_in[8];
  const float* te   = (const float*)d_in[9];
  const float* Wih0 = (const float*)d_in[10];
  const float* Whh0 = (const float*)d_in[11];
  const float* bih0 = (const float*)d_in[12];
  const float* bhh0 = (const float*)d_in[13];
  const float* Wih1 = (const float*)d_in[14];
  const float* Whh1 = (const float*)d_in[15];
  const float* bih1 = (const float*)d_in[16];
  const float* bhh1 = (const float*)d_in[17];
  const float* Wout = (const float*)d_in[18];
  const float* bout = (const float*)d_in[19];

  char* w = (char*)d_ws;
  auto take = [&](size_t bytes) -> char* {
    char* p = w;
    w += (bytes + 255) & ~(size_t)255;
    return p;
  };
  _Float16* x0    = (_Float16*)take((size_t)kB * kT * kIN0 * 2);   // 20 MB
  _Float16* h1    = (_Float16*)take((size_t)kB * kT * kH * 2);     // 80 MB
  _Float16* h2    = (_Float16*)take((size_t)kB * kH * 2);          //  8 MB
  _Float16* Wcat0 = (_Float16*)take((size_t)kG * (kH + kIN0) * 2);
  _Float16* Wcat1 = (_Float16*)take((size_t)kG * (kH + kH) * 2);
  _Float16* WoutP = (_Float16*)take((size_t)kNOUTP * kH * 2);
  float*    b0    = (float*)take((size_t)kG * 4);
  float*    b1    = (float*)take((size_t)kG * 4);
  float*    bo    = (float*)take((size_t)kNOUTP * 4);

  const int TPB = 256;
  prep_wcat_k<<<(kG * (kH + kIN0) + TPB - 1) / TPB, TPB, 0, stream>>>(
      Whh0, Wih0, kINr, kIN0, bih0, bhh0, Wcat0, b0);
  prep_wcat_k<<<(kG * (kH + kH) + TPB - 1) / TPB, TPB, 0, stream>>>(
      Whh1, Wih1, kH, kH, bih1, bhh1, Wcat1, b1);
  prep_wout_k<<<(kNOUTP * kH + TPB - 1) / TPB, TPB, 0, stream>>>(Wout, bout, WoutP, bo);
  gather_embed_k<<<(kB * kT + TPB - 1) / TPB, TPB, 0, stream>>>(
      car, reg, poi, wk, tm, ce, re, pe, we, te, x0);

  lstm_layer_k<kIN0, true ><<<kB / kMT, TPB, 0, stream>>>(x0, Wcat0, b0, h1);
  lstm_layer_k<kH,   false><<<kB / kMT, TPB, 0, stream>>>(h1, Wcat1, b1, h2);
  out_gemm_k<<<kB / 16, TPB, 0, stream>>>(h2, WoutP, bo, (float*)d_out);
}
